// ETSformerBackbone_73753178406955
// MI455X (gfx1250) — matrix operand order repaired and checked by reading, not compiled, not run on hardware
//
#include <hip/hip_runtime.h>
#include <hip/hip_bf16.h>
#include <math.h>

// ---------------- problem constants ----------------
constexpr int B = 64, L = 512, P = 96, V = 7, D = 512, F = 2048, E = 3, K = 4;
constexpr int M_ROWS = B * L;                 // 32768 GEMM rows
constexpr float PI_F = 3.14159265358979323846f;

// ---------------- helpers ----------------
__device__ __forceinline__ unsigned short f32_to_bf16(float f) {
    unsigned int u = __float_as_uint(f);
    unsigned int r = u + 0x7fffu + ((u >> 16) & 1u);   // round-to-nearest-even
    return (unsigned short)(r >> 16);
}

typedef __attribute__((ext_vector_type(16))) __bf16 v16bf;
typedef __attribute__((ext_vector_type(8)))  float  v8f;
typedef int v4i_ __attribute__((vector_size(4 * sizeof(int))));   // matches builtin param type

// CDNA5 async global->LDS copy path (ASYNCcnt-tracked), with sync fallback.
#if defined(__gfx1250__) && __has_builtin(__builtin_amdgcn_global_load_async_to_lds_b128) && \
    __has_builtin(__builtin_amdgcn_s_wait_asynccnt)
#define USE_ASYNC_LDS 1
#else
#define USE_ASYNC_LDS 0
#endif

// ================= instance-norm stats =================
__global__ void k_stats(const float* __restrict__ x, float* __restrict__ means,
                        float* __restrict__ stdev) {
    int idx = blockIdx.x * blockDim.x + threadIdx.x;   // b*V + v
    if (idx >= B * V) return;
    int b = idx / V, v = idx % V;
    float s = 0.f;
    for (int t = 0; t < L; ++t) s += x[((size_t)b * L + t) * V + v];
    float m = s / (float)L;
    float s2 = 0.f;
    for (int t = 0; t < L; ++t) {
        float c = x[((size_t)b * L + t) * V + v] - m;
        s2 += c * c;
    }
    means[idx] = m;
    stdev[idx] = sqrtf(s2 / (float)L + 1e-5f);
}

// ================= normalize + embed =================
__global__ void k_embed(const float* __restrict__ x, const float* __restrict__ means,
                        const float* __restrict__ stdev, const float* __restrict__ Wemb,
                        const float* __restrict__ bemb, float* __restrict__ hidden) {
    size_t idx = (size_t)blockIdx.x * blockDim.x + threadIdx.x;   // b*L*D + t*D + d
    if (idx >= (size_t)B * L * D) return;
    int d = (int)(idx % D);
    size_t bl = idx / D;
    int b = (int)(bl / L);
    float acc = bemb[d];
#pragma unroll
    for (int v = 0; v < V; ++v) {
        float xn = (x[bl * V + v] - means[b * V + v]) / stdev[b * V + v];
        acc += xn * Wemb[v * D + d];
    }
    hidden[idx] = acc;
}

// ================= dual EMA scan + Fourier coefficients =================
__global__ void k_scan(const float* __restrict__ hidden, const float* __restrict__ a_lvl,
                       const float* __restrict__ a_gr, float* __restrict__ level,
                       float* __restrict__ growth, float* __restrict__ coef) {
    int idx = blockIdx.x * blockDim.x + threadIdx.x;   // b*D + d
    if (idx >= B * D) return;
    int b = idx / D, d = idx % D;
    float al = 1.f / (1.f + __expf(-a_lvl[d]));
    float ag = 1.f / (1.f + __expf(-a_gr[d]));
    float s = 0.f, g = 0.f, hprev = 0.f;
    float cr[K], ci[K];
#pragma unroll
    for (int k = 0; k < K; ++k) { cr[k] = 0.f; ci[k] = 0.f; }
    const float w0 = -2.f * PI_F / (float)L;
    for (int t = 0; t < L; ++t) {
        float h = hidden[((size_t)b * L + t) * D + d];
        if (t == 0) { s = h; g = 0.f; }
        else {
            s = al * h + (1.f - al) * s;
            float diff = h - hprev;
            g = ag * diff + (1.f - ag) * g;
        }
        hprev = h;
        level[((size_t)b * L + t) * D + d]  = s;
        growth[((size_t)b * L + t) * D + d] = g;
        float r = h - s;
#pragma unroll
        for (int k = 0; k < K; ++k) {
            float sn, cs;
            __sincosf(w0 * (float)(k * t), &sn, &cs);
            cr[k] += r * cs;
            ci[k] += r * sn;
        }
    }
#pragma unroll
    for (int k = 0; k < K; ++k) {
        coef[(((size_t)b * K + k) * D + d) * 2 + 0] = cr[k];
        coef[(((size_t)b * K + k) * D + d) * 2 + 1] = ci[k];
    }
}

// ================= seasonal reconstruction + fuse =================
__global__ void k_fuse(const float* __restrict__ level, const float* __restrict__ growth,
                       const float* __restrict__ coef, const float* __restrict__ wr,
                       const float* __restrict__ wi, float* __restrict__ seasonal,
                       float* __restrict__ fused, unsigned short* __restrict__ fused_bf) {
    size_t idx = (size_t)blockIdx.x * blockDim.x + threadIdx.x;
    if (idx >= (size_t)B * L * D) return;
    int d = (int)(idx % D);
    int t = (int)((idx / D) % L);
    int b = (int)(idx / ((size_t)D * L));
    const float w0 = 2.f * PI_F / (float)L;
    float acc = 0.f;
#pragma unroll
    for (int k = 0; k < K; ++k) {
        float crr = coef[(((size_t)b * K + k) * D + d) * 2 + 0];
        float cii = coef[(((size_t)b * K + k) * D + d) * 2 + 1];
        float wrr = wr[k * D + d], wii = wi[k * D + d];
        float yr = crr * wrr - cii * wii;
        float yi = crr * wii + cii * wrr;
        float sn, cs;
        __sincosf(w0 * (float)(k * t), &sn, &cs);
        float re = yr * cs - yi * sn;
        acc += (k == 0) ? re : 2.f * re;        // irfft with only low-K bins live
    }
    float seas = acc / (float)L;
    float f = level[idx] + growth[idx] + seas;
    seasonal[idx] = seas;
    fused[idx]    = f;
    fused_bf[idx] = f32_to_bf16(f);
}

// ================= transpose + convert f32 -> bf16 =================
__global__ void k_transpose_bf16(const float* __restrict__ src, unsigned short* __restrict__ dst,
                                 int R, int C) {
    size_t idx = (size_t)blockIdx.x * blockDim.x + threadIdx.x;
    if (idx >= (size_t)R * C) return;
    int r = (int)(idx % R);
    int c = (int)(idx / R);
    dst[idx] = f32_to_bf16(src[(size_t)r * C + c]);
}

// ================= bf16 WMMA GEMM =================
// C(MxN) = A(MxK) * Bt(NxK)^T ; EPI==1: +bias, gelu, bf16 out; EPI==2: +bias +resid, f32 out
constexpr int TM = 128, TN = 128, TK = 64;

template <int EPI>
__global__ __launch_bounds__(256) void k_gemm(const unsigned short* __restrict__ A,
                                              const unsigned short* __restrict__ Bt,
                                              const float* __restrict__ bias,
                                              const float* __restrict__ resid,
                                              unsigned short* __restrict__ outb,
                                              float* __restrict__ outf,
                                              int M, int N, int Kd) {
    __shared__ unsigned short lA[2][TM * TK];   // double-buffered 128 x 64 bf16 (16KB each)
    __shared__ unsigned short lB[2][TN * TK];

    const int tid  = threadIdx.x;
    const int lane = tid & 31;
    const int wave = tid >> 5;        // 8 waves
    const int wm   = wave >> 1;       // 0..3  (M blocks of 32)
    const int wn   = wave & 1;        // 0..1  (N blocks of 64)
    const int h    = lane >> 4;       // lane half
    const int l16  = lane & 15;

    const int bm = blockIdx.x * TM;
    const int bn = blockIdx.y * TN;

    v8f acc[2][4];
    const v8f vzero = {};
#pragma unroll
    for (int i = 0; i < 2; ++i)
#pragma unroll
        for (int j = 0; j < 4; ++j) acc[i][j] = vzero;

    // copy mapping: 2 threads per 64-k row; each thread moves 64B (4 x b128) of A and of B
    const int gr = tid >> 1;
    const int gk = (tid & 1) * 32;    // element offset 0 or 32

    auto copy_tiles = [&](int buf, int k0) {
        const unsigned short* gA = A  + (size_t)(bm + gr) * Kd + k0 + gk;
        const unsigned short* gB = Bt + (size_t)(bn + gr) * Kd + k0 + gk;
        unsigned short* sA = &lA[buf][gr * TK + gk];
        unsigned short* sB = &lB[buf][gr * TK + gk];
#if USE_ASYNC_LDS
#pragma unroll
        for (int c = 0; c < 4; ++c)
            __builtin_amdgcn_global_load_async_to_lds_b128(
                (v4i_*)(gA + 8 * c), (v4i_*)(sA + 8 * c), 0, 0);
#pragma unroll
        for (int c = 0; c < 4; ++c)
            __builtin_amdgcn_global_load_async_to_lds_b128(
                (v4i_*)(gB + 8 * c), (v4i_*)(sB + 8 * c), 0, 0);
#else
        uint4 ra[4], rb[4];
#pragma unroll
        for (int c = 0; c < 4; ++c) ra[c] = *(const uint4*)(gA + 8 * c);
#pragma unroll
        for (int c = 0; c < 4; ++c) rb[c] = *(const uint4*)(gB + 8 * c);
#pragma unroll
        for (int c = 0; c < 4; ++c) *(uint4*)(sA + 8 * c) = ra[c];
#pragma unroll
        for (int c = 0; c < 4; ++c) *(uint4*)(sB + 8 * c) = rb[c];
#endif
    };

    const int nk = Kd / TK;
    copy_tiles(0, 0);   // prefetch first tile

    for (int kt = 0; kt < nk; ++kt) {
        const int cur = kt & 1;
        if (kt + 1 < nk) {
            copy_tiles(1 - cur, (kt + 1) * TK);   // prefetch next tile into other buffer
#if USE_ASYNC_LDS
            __builtin_amdgcn_s_wait_asynccnt(8);  // current tile's 8 async copies done
#endif
        } else {
#if USE_ASYNC_LDS
            __builtin_amdgcn_s_wait_asynccnt(0);
#endif
        }
        __syncthreads();   // all waves' copies for `cur` visible

#pragma unroll
        for (int s = 0; s < 2; ++s) {   // two K=32 WMMA steps per 64-deep tile
            union FB { uint4 u[2]; v16bf v; };
            FB af[2], bf2[4];
#pragma unroll
            for (int i = 0; i < 2; ++i) {
                int row = wm * 32 + i * 16 + l16;
                // A layout (ISA 16-bit A 16x32): K = [8h, 8h+8) then [16+8h, 24+8h)
                af[i].u[0] = *(const uint4*)(&lA[cur][row * TK + s * 32 + 8 * h]);
                af[i].u[1] = *(const uint4*)(&lA[cur][row * TK + s * 32 + 16 + 8 * h]);
            }
#pragma unroll
            for (int j = 0; j < 4; ++j) {
                int col = wn * 64 + j * 16 + l16;
                // B layout (same as A): K = [8h, 8h+8) then [16+8h, 24+8h)
                bf2[j].u[0] = *(const uint4*)(&lB[cur][col * TK + s * 32 + 8 * h]);
                bf2[j].u[1] = *(const uint4*)(&lB[cur][col * TK + s * 32 + 16 + 8 * h]);
            }
#pragma unroll
            for (int i = 0; i < 2; ++i)
#pragma unroll
                for (int j = 0; j < 4; ++j)
                    acc[i][j] = __builtin_amdgcn_wmma_f32_16x16x32_bf16(
                        false, af[i].v, false, bf2[j].v, (short)0, acc[i][j], false, false);
        }
        __syncthreads();   // everyone done reading `cur` before it is overwritten
    }

    // epilogue; C layout: VGPR r -> M = r + 8*h, N = lane&15
#pragma unroll
    for (int i = 0; i < 2; ++i) {
#pragma unroll
        for (int j = 0; j < 4; ++j) {
            int ncol = bn + wn * 64 + j * 16 + l16;
            float bv = bias[ncol];
#pragma unroll
            for (int r = 0; r < 8; ++r) {
                int mrow = bm + wm * 32 + i * 16 + r + 8 * h;
                float v = acc[i][j][r] + bv;
                if (EPI == 1) {
                    float t = tanhf(0.7978845608028654f * (v + 0.044715f * v * v * v));
                    float gl = 0.5f * v * (1.f + t);
                    outb[(size_t)mrow * N + ncol] = f32_to_bf16(gl);
                } else {
                    v += resid[(size_t)mrow * N + ncol];
                    outf[(size_t)mrow * N + ncol] = v;
                }
            }
        }
    }
}

// ================= layernorm (in-place capable; wave per row) =================
__global__ __launch_bounds__(256) void k_ln(const float* __restrict__ pre,
                                            const float* __restrict__ gam,
                                            const float* __restrict__ bet,
                                            float* __restrict__ out) {
    int row  = blockIdx.x * 8 + (threadIdx.x >> 5);
    int lane = threadIdx.x & 31;
    const float* x = pre + (size_t)row * D;
    float vals[16];
    float m = 0.f;
#pragma unroll
    for (int i = 0; i < 16; ++i) { vals[i] = x[lane + i * 32]; m += vals[i]; }
#pragma unroll
    for (int off = 16; off > 0; off >>= 1) m += __shfl_xor(m, off, 32);
    m /= (float)D;
    float var = 0.f;
#pragma unroll
    for (int i = 0; i < 16; ++i) { float c = vals[i] - m; var += c * c; }
#pragma unroll
    for (int off = 16; off > 0; off >>= 1) var += __shfl_xor(var, off, 32);
    var /= (float)D;
    float rs = rsqrtf(var + 1e-5f);
#pragma unroll
    for (int i = 0; i < 16; ++i) {
        int d = lane + i * 32;
        out[(size_t)row * D + d] = (vals[i] - m) * rs * gam[d] + bet[d];
    }
}

// ================= heads =================
__global__ void k_head_lg(const float* __restrict__ level, const float* __restrict__ growth,
                          const float* __restrict__ Wl, const float* __restrict__ bl,
                          const float* __restrict__ Wg, const float* __restrict__ bg,
                          float* __restrict__ lb, float* __restrict__ gs) {
    int idx = blockIdx.x * blockDim.x + threadIdx.x;   // b*V + v
    if (idx >= B * V) return;
    int b = idx / V, v = idx % V;
    const float* lv = level  + ((size_t)b * L + (L - 1)) * D;
    const float* gr = growth + ((size_t)b * L + (L - 1)) * D;
    float a = bl[v], c = bg[v];
    for (int d = 0; d < D; ++d) {
        a += lv[d] * Wl[d * V + v];
        c += gr[d] * Wg[d * V + v];
    }
    lb[idx] = a;
    gs[idx] = c;
}

__global__ void k_season_time(const float* __restrict__ seasonal, const float* __restrict__ Wt,
                              const float* __restrict__ bt, float* __restrict__ sfc1) {
    size_t idx = (size_t)blockIdx.x * blockDim.x + threadIdx.x;   // b*P*D + p*D + d
    if (idx >= (size_t)B * P * D) return;
    int d = (int)(idx % D);
    int p = (int)((idx / D) % P);
    int b = (int)(idx / ((size_t)D * P));
    float acc = bt[p];
    for (int l = 0; l < L; ++l)
        acc += seasonal[((size_t)b * L + l) * D + d] * Wt[l * P + p];
    sfc1[idx] = acc;
}

__global__ void k_final(const float* __restrict__ sfc1, const float* __restrict__ Wf,
                        const float* __restrict__ bf_, const float* __restrict__ lb,
                        const float* __restrict__ gs, const float* __restrict__ means,
                        const float* __restrict__ stdev, float* __restrict__ out) {
    int idx = blockIdx.x * blockDim.x + threadIdx.x;   // b*P*V + p*V + v
    if (idx >= B * P * V) return;
    int v = idx % V;
    int p = (idx / V) % P;
    int b = idx / (V * P);
    float acc = bf_[v];
    const float* s = sfc1 + ((size_t)b * P + p) * D;
    for (int d = 0; d < D; ++d) acc += s[d] * Wf[d * V + v];
    float dec = lb[b * V + v] + gs[b * V + v] * (float)(p + 1) + acc;
    out[idx] = dec * stdev[b * V + v] + means[b * V + v];
}

// ================= host launcher =================
extern "C" void kernel_launch(void* const* d_in, const int* in_sizes, int n_in,
                              void* d_out, int out_size, void* d_ws, size_t ws_size,
                              hipStream_t stream) {
    (void)in_sizes; (void)n_in; (void)out_size; (void)ws_size;
    const float* x      = (const float*)d_in[0];
    const float* W_emb  = (const float*)d_in[1];
    const float* b_emb  = (const float*)d_in[2];
    const float* a_lvl  = (const float*)d_in[3];
    const float* a_gr   = (const float*)d_in[4];
    const float* wr     = (const float*)d_in[5];
    const float* wi     = (const float*)d_in[6];
    const float* W1     = (const float*)d_in[7];
    const float* b1     = (const float*)d_in[8];
    const float* W2     = (const float*)d_in[9];
    const float* b2     = (const float*)d_in[10];
    const float* ln_g   = (const float*)d_in[11];
    const float* ln_b   = (const float*)d_in[12];
    const float* Wl     = (const float*)d_in[13];
    const float* bl     = (const float*)d_in[14];
    const float* Wg     = (const float*)d_in[15];
    const float* bg     = (const float*)d_in[16];
    const float* Wt     = (const float*)d_in[17];
    const float* bt     = (const float*)d_in[18];
    const float* Wf     = (const float*)d_in[19];
    const float* bf_    = (const float*)d_in[20];
    float* out = (float*)d_out;

    // bump-allocate scratch
    char* base = (char*)d_ws;
    size_t off = 0;
    auto alloc = [&](size_t bytes) -> char* {
        char* p = base + off;
        off = (off + bytes + 255) & ~(size_t)255;
        return p;
    };
    const size_t BLD = (size_t)B * L * D;
    float* means   = (float*)alloc((size_t)B * V * 4);
    float* stdev   = (float*)alloc((size_t)B * V * 4);
    float* hidden  = (float*)alloc(BLD * 4);             // doubles as pre-LN buffer
    float* level   = (float*)alloc(BLD * 4);
    float* growth  = (float*)alloc(BLD * 4);
    float* seasonal= (float*)alloc(BLD * 4);
    float* fused   = (float*)alloc(BLD * 4);
    float* coef    = (float*)alloc((size_t)B * K * D * 2 * 4);
    unsigned short* fused_bf = (unsigned short*)alloc(BLD * 2);
    unsigned short* act      = (unsigned short*)alloc((size_t)B * L * F * 2);
    unsigned short* W1t      = (unsigned short*)alloc((size_t)F * D * 2);
    unsigned short* W2t      = (unsigned short*)alloc((size_t)D * F * 2);
    float* lb   = (float*)alloc((size_t)B * V * 4);
    float* gs   = (float*)alloc((size_t)B * V * 4);
    float* sfc1 = (float*)alloc((size_t)B * P * D * 4);

    const int T256 = 256;
    k_stats<<<(B * V + T256 - 1) / T256, T256, 0, stream>>>(x, means, stdev);
    k_embed<<<(int)((BLD + T256 - 1) / T256), T256, 0, stream>>>(x, means, stdev, W_emb, b_emb, hidden);

    for (int e = 0; e < E; ++e) {
        k_scan<<<(B * D + T256 - 1) / T256, T256, 0, stream>>>(
            hidden, a_lvl + (size_t)e * D, a_gr + (size_t)e * D, level, growth, coef);
        k_fuse<<<(int)((BLD + T256 - 1) / T256), T256, 0, stream>>>(
            level, growth, coef, wr + (size_t)e * K * D, wi + (size_t)e * K * D,
            seasonal, fused, fused_bf);
        k_transpose_bf16<<<(int)(((size_t)D * F + T256 - 1) / T256), T256, 0, stream>>>(
            W1 + (size_t)e * D * F, W1t, D, F);   // W1t: (F x D)
        k_transpose_bf16<<<(int)(((size_t)F * D + T256 - 1) / T256), T256, 0, stream>>>(
            W2 + (size_t)e * F * D, W2t, F, D);   // W2t: (D x F)
        k_gemm<1><<<dim3(M_ROWS / TM, F / TN), 256, 0, stream>>>(
            fused_bf, W1t, b1 + (size_t)e * F, nullptr, act, nullptr, M_ROWS, F, D);
        k_gemm<2><<<dim3(M_ROWS / TM, D / TN), 256, 0, stream>>>(
            act, W2t, b2 + (size_t)e * D, fused, nullptr, hidden, M_ROWS, D, F);
        k_ln<<<M_ROWS / 8, 256, 0, stream>>>(hidden, ln_g + (size_t)e * D, ln_b + (size_t)e * D, hidden);
    }

    k_head_lg<<<(B * V + T256 - 1) / T256, T256, 0, stream>>>(level, growth, Wl, bl, Wg, bg, lb, gs);
    k_season_time<<<(int)(((size_t)B * P * D + T256 - 1) / T256), T256, 0, stream>>>(seasonal, Wt, bt, sfc1);
    k_final<<<(B * P * V + T256 - 1) / T256, T256, 0, stream>>>(sfc1, Wf, bf_, lb, gs, means, stdev, out);
}
